// AttentionHead_17042430231165
// MI455X (gfx1250) — matrix operand order repaired and checked by reading, not compiled, not run on hardware
//
#include <hip/hip_runtime.h>
#include <hip/hip_bf16.h>
#include <math.h>

#define WSZ   32
#define INSZ  8192
#define KEYSZ 256
#define VALSZ 8192
#define SCALE (1.0f / 256.0f)

typedef __attribute__((ext_vector_type(16))) __bf16 v16bf;
typedef __attribute__((ext_vector_type(8)))  float  v8f;
typedef __attribute__((ext_vector_type(4)))  float  v4f;
typedef __attribute__((ext_vector_type(2)))  float  v2f;

// ---------------------------------------------------------------------------
// A fragment: rows [m0, m0+16) of row-major src (leading dim ld), K chunk
// [kc, kc+32). Lanes 0-15 hold K 0..15, lanes 16-31 hold K 16..31 (matches B);
// VGPR v packs two consecutive K values.
// Each lane -> one run of 16 contiguous floats -> 4x global_load_b128 + cvt.
// ---------------------------------------------------------------------------
__device__ __forceinline__ v16bf load_frag_A(const float* __restrict__ src,
                                             int ld, int m0, int kc) {
    const int lane = threadIdx.x & 31;
    const int half = lane >> 4;
    const float* p = src + (size_t)(m0 + (lane & 15)) * ld + kc + half * 16;
    v4f f0 = *(const v4f*)(p + 0);
    v4f f1 = *(const v4f*)(p + 4);
    v4f f2 = *(const v4f*)(p + 8);
    v4f f3 = *(const v4f*)(p + 12);
    v16bf r;
#pragma unroll
    for (int e = 0; e < 4; ++e) {
        r[e]      = (__bf16)f0[e];
        r[4 + e]  = (__bf16)f1[e];
        r[8 + e]  = (__bf16)f2[e];
        r[12 + e] = (__bf16)f3[e];
    }
    return r;
}

// ---------------------------------------------------------------------------
// B fragment: B[k][n] = w[n0+n][kc+k] (w row-major, leading dim ld), 32x16.
// Lane n = lane&15; lanes 0-15 hold K 0..15, lanes 16-31 hold K 16..31
// (VGPR v packs K=2v,2v+1). Each lane: 16 contiguous floats from one w row.
// ---------------------------------------------------------------------------
__device__ __forceinline__ v16bf load_frag_B(const float* __restrict__ w,
                                             int ld, int n0, int kc) {
    const int lane = threadIdx.x & 31;
    const int half = lane >> 4;
    const float* p = w + (size_t)(n0 + (lane & 15)) * ld + kc + half * 16;
    v4f f0 = *(const v4f*)(p + 0);
    v4f f1 = *(const v4f*)(p + 4);
    v4f f2 = *(const v4f*)(p + 8);
    v4f f3 = *(const v4f*)(p + 12);
    v16bf r;
#pragma unroll
    for (int e = 0; e < 4; ++e) {
        r[e]      = (__bf16)f0[e];
        r[4 + e]  = (__bf16)f1[e];
        r[8 + e]  = (__bf16)f2[e];
        r[12 + e] = (__bf16)f3[e];
    }
    return r;
}

// ===========================================================================
// Kernel 1: q/k projections (no bias; bias folded into kernel 2).
// 64 blocks: {khalf, mat(k|q), n-tile}; 2 waves/block = M tiles 0-15 / 16-31.
// ws layout (floats): k0 | k1 | q0 | q1, each WSZ*KEYSZ.
// ===========================================================================
__global__ void __launch_bounds__(64)
qk_proj_kernel(const float* __restrict__ x, const float* __restrict__ wk,
               const float* __restrict__ wq, float* __restrict__ ws) {
    const int bx    = blockIdx.x;      // 0..63
    const int khalf = bx >> 5;
    const int rem   = bx & 31;
    const int mat   = rem >> 4;        // 0 = k, 1 = q
    const int n0    = (rem & 15) * 16;
    const int wave  = threadIdx.x >> 5;
    const int m0    = wave * 16;
    const float* w  = mat ? wq : wk;
    float* dst = ws + (size_t)(mat * 2 + khalf) * (WSZ * KEYSZ);

    v8f acc = {};
    const int kbeg = khalf * (INSZ / 2);
    const int kend = kbeg + INSZ / 2;
#pragma unroll 2
    for (int kc = kbeg; kc < kend; kc += 32) {
        v16bf a = load_frag_A(x, INSZ, m0, kc);
        v16bf b = load_frag_B(w, INSZ, n0, kc);
        acc = __builtin_amdgcn_wmma_f32_16x16x32_bf16(false, a, false, b,
                                                      (short)0, acc, false, false);
    }

    const int lane = threadIdx.x & 31;
    const int half = lane >> 4;
    const int n    = n0 + (lane & 15);
#pragma unroll
    for (int r = 0; r < 8; ++r) {
        const int m = m0 + r + half * 8;          // C layout: VGPR r -> M=r / 8+r
        dst[m * KEYSZ + n] = acc[r];
    }
}

// ===========================================================================
// Kernel 2: scores = min(q k^T * SCALE, tri); row softmax -> a [32x32].
// One block 32x32; each row is exactly one wave32 -> shuffle reductions.
// Combines the two K-half partials and adds biases on the fly.
// ===========================================================================
__global__ void __launch_bounds__(1024)
softmax_kernel(const float* __restrict__ ws_qk, const float* __restrict__ kb,
               const float* __restrict__ qb, const float* __restrict__ tri,
               float* __restrict__ a_out) {
    const int j = threadIdx.x;   // key index, lane within wave
    const int i = threadIdx.y;   // query index (one wave per row)
    const float* k0 = ws_qk;
    const float* k1 = ws_qk + 1 * WSZ * KEYSZ;
    const float* q0 = ws_qk + 2 * WSZ * KEYSZ;
    const float* q1 = ws_qk + 3 * WSZ * KEYSZ;

    float s = 0.0f;
    for (int t = 0; t < KEYSZ; ++t) {
        float qv = q0[i * KEYSZ + t] + q1[i * KEYSZ + t] + qb[t];
        float kv = k0[j * KEYSZ + t] + k1[j * KEYSZ + t] + kb[t];
        s = fmaf(qv, kv, s);
    }
    s = fminf(s * SCALE, tri[i * WSZ + j]);

    float m = s;
#pragma unroll
    for (int off = 16; off >= 1; off >>= 1)
        m = fmaxf(m, __shfl_xor(m, off, 32));
    float e = __expf(s - m);
    float sum = e;
#pragma unroll
    for (int off = 16; off >= 1; off >>= 1)
        sum += __shfl_xor(sum, off, 32);

    a_out[i * WSZ + j] = e / sum;
}

// ===========================================================================
// Kernel 3: the 256 MB streamer.  out[:, tile] = a @ (x @ wv^T + b).
// 128 blocks x 4 waves; each wave owns 16 output columns.  bf16 WMMA over
// K=8192 builds the v-tile in registers; tiny a@v epilogue uses fp32 WMMA
// (16x16x4) through LDS so post-softmax accumulation stays full precision.
// ===========================================================================
__global__ void __launch_bounds__(128)
v_attn_kernel(const float* __restrict__ x, const float* __restrict__ wv,
              const float* __restrict__ vb, const float* __restrict__ a_mat,
              float* __restrict__ out) {
    __shared__ float a_s[WSZ * WSZ];
    __shared__ float v_s[4][WSZ * 16];

    const int tid  = threadIdx.x;
    const int wave = tid >> 5;
    const int lane = tid & 31;
    const int half = lane >> 4;
    const int n0   = blockIdx.x * 64 + wave * 16;

    for (int t = tid; t < WSZ * WSZ; t += 128) a_s[t] = a_mat[t];

    v8f acc0 = {}, acc1 = {};
#pragma unroll 2
    for (int kc = 0; kc < INSZ; kc += 32) {
        v16bf b  = load_frag_B(wv, INSZ, n0, kc);
        v16bf a0 = load_frag_A(x, INSZ, 0, kc);
        v16bf a1 = load_frag_A(x, INSZ, 16, kc);
        acc0 = __builtin_amdgcn_wmma_f32_16x16x32_bf16(false, a0, false, b,
                                                       (short)0, acc0, false, false);
        acc1 = __builtin_amdgcn_wmma_f32_16x16x32_bf16(false, a1, false, b,
                                                       (short)0, acc1, false, false);
    }

    // v includes bias; rows of a sum to 1, so adding bias pre-(a@v) is exact.
    const float bias = vb[n0 + (lane & 15)];
#pragma unroll
    for (int r = 0; r < 8; ++r) { acc0[r] += bias; acc1[r] += bias; }

    // Spill v-tile (32 rows x 16 cols) to LDS in plain [j][n] row-major form.
    float* vs = v_s[wave];
#pragma unroll
    for (int r = 0; r < 8; ++r) {
        vs[(r + half * 8) * 16 + (lane & 15)]        = acc0[r];
        vs[(16 + r + half * 8) * 16 + (lane & 15)]   = acc1[r];
    }
    __syncthreads();   // fences v_s (per-wave) and a_s (block-wide)

    // a @ v via fp32 WMMA 16x16x4, K = 32 in 8 steps.
    // A 16x4 f32 layout: lane M = lane&15; VGPR0 K = jc+2*half, VGPR1 K = +1.
    v8f o0 = {}, o1 = {};
#pragma unroll
    for (int jc = 0; jc < WSZ; jc += 4) {
        const int kA = jc + half * 2;
        v2f af0, af1, bf;
        af0[0] = a_s[(lane & 15) * WSZ + kA];
        af0[1] = a_s[(lane & 15) * WSZ + kA + 1];
        af1[0] = a_s[(16 + (lane & 15)) * WSZ + kA];
        af1[1] = a_s[(16 + (lane & 15)) * WSZ + kA + 1];
        bf[0]  = vs[kA * 16 + (lane & 15)];
        bf[1]  = vs[(kA + 1) * 16 + (lane & 15)];
        o0 = __builtin_amdgcn_wmma_f32_16x16x4_f32(false, af0, false, bf,
                                                   (short)0, o0, false, false);
        o1 = __builtin_amdgcn_wmma_f32_16x16x4_f32(false, af1, false, bf,
                                                   (short)0, o1, false, false);
    }

    const int n = n0 + (lane & 15);
#pragma unroll
    for (int r = 0; r < 8; ++r) {
        out[(size_t)(r + half * 8) * VALSZ + n]      = o0[r];
        out[(size_t)(16 + r + half * 8) * VALSZ + n] = o1[r];
    }
}

// ===========================================================================
extern "C" void kernel_launch(void* const* d_in, const int* in_sizes, int n_in,
                              void* d_out, int out_size, void* d_ws, size_t ws_size,
                              hipStream_t stream) {
    const float* x    = (const float*)d_in[0];
    const float* wk_w = (const float*)d_in[1];
    const float* wk_b = (const float*)d_in[2];
    const float* wq_w = (const float*)d_in[3];
    const float* wq_b = (const float*)d_in[4];
    const float* wv_w = (const float*)d_in[5];
    const float* wv_b = (const float*)d_in[6];
    const float* tri  = (const float*)d_in[7];
    float* out = (float*)d_out;
    float* ws  = (float*)d_ws;                 // 4*32*256 + 32*32 floats used
    float* a_mat = ws + 4 * WSZ * KEYSZ;

    qk_proj_kernel<<<64, 64, 0, stream>>>(x, wk_w, wq_w, ws);
    dim3 b2(32, 32);
    softmax_kernel<<<1, b2, 0, stream>>>(ws, wk_b, wq_b, tri, a_mat);
    v_attn_kernel<<<VALSZ / 64, 128, 0, stream>>>(x, wv_w, wv_b, a_mat, out);
}
